// Decoder_2224793060068
// MI455X (gfx1250) — matrix operand order repaired and checked by reading, not compiled, not run on hardware
//
#include <hip/hip_runtime.h>

typedef unsigned int   u32;
typedef unsigned short u16;
typedef __attribute__((ext_vector_type(16))) __bf16 v16bf;
typedef __attribute__((ext_vector_type(8)))  float  v8f;
typedef __attribute__((ext_vector_type(4)))  u32    u32x4;
typedef __attribute__((ext_vector_type(8)))  int    i32x8;
typedef __attribute__((ext_vector_type(4)))  int    i32x4;

#define BB 16
#define TT 128
#define SS 128
#define VV 32000
#define EE 256
#define FF 512
#define KZ 1280   // combined LSTM K: (E + F) + F = 768 + 512
#define NZ 2048   // 4*F gate columns

#if __has_builtin(__builtin_amdgcn_tensor_load_to_lds)
#define HAVE_TDM 1
#else
#define HAVE_TDM 0
#endif

union ABFrag { v16bf v; u32 u[8]; };

__device__ __forceinline__ float bf2f(u16 h) {
  u32 u = ((u32)h) << 16; float f; __builtin_memcpy(&f, &u, 4); return f;
}
__device__ __forceinline__ u16 f2bf(float x) {
  u32 u; __builtin_memcpy(&u, &x, 4);
  u32 r = u + 0x7fffu + ((u >> 16) & 1u);
  return (u16)(r >> 16);
}
__device__ __forceinline__ float sigmoidf(float x) { return 1.f / (1.f + __expf(-x)); }

__device__ __forceinline__ v8f wmma_bf16(v16bf a, v16bf b, v8f c) {
  // D = A(16x32 bf16) * B(32x16 bf16) + C(16x16 f32)
  return __builtin_amdgcn_wmma_f32_16x16x32_bf16(false, a, false, b, (short)0, c, false, false);
}

// A fragment (16x32, M x K): row m = lane&15; K pairs {0..7,16..23} (lanes 0-15)
// or {8..15,24..31} (lanes 16-31) per ISA 7.12.2. Source row-major, leading dim ld.
__device__ __forceinline__ v16bf load_a_frag(const u16* base, int ld, int k0, int lane) {
  ABFrag r;
  const u16* p = base + (size_t)(lane & 15) * ld + k0 + ((lane >> 4) & 1) * 8;
#pragma unroll
  for (int v = 0; v < 4; ++v) r.u[v] = *(const u32*)(p + 2 * v);
#pragma unroll
  for (int v = 0; v < 4; ++v) r.u[4 + v] = *(const u32*)(p + 16 + 2 * v);
  return r.v;
}

// B fragment (32x16, K x N) sourced from B^T stored row-major [N][K]:
// column n = n_base + (lane&15); K pairs {0..7,16..23} (lanes 0-15) or {8..15,24..31} (lanes 16-31).
__device__ __forceinline__ v16bf load_b_frag(const u16* bt, int ld, int n_base, int k0, int lane) {
  ABFrag r;
  const u16* p = bt + (size_t)(n_base + (lane & 15)) * ld + k0 + ((lane >> 4) & 1) * 8;
#pragma unroll
  for (int v = 0; v < 4; ++v) r.u[v] = *(const u32*)(p + 2 * v);
#pragma unroll
  for (int v = 0; v < 4; ++v) r.u[4 + v] = *(const u32*)(p + 16 + 2 * v);
  return r.v;
}

// ---------------- Tensor Data Mover: 2D bf16 tile global -> LDS ----------------
__device__ __forceinline__ void tdm_load_tile(const void* gsrc, u32 lds_byte,
                                              u32 tile_k, u32 tile_rows, u32 row_stride_elems) {
#if HAVE_TDM
  unsigned long long ga = (unsigned long long)gsrc;
  // D# group0: count=1, lds_addr, global_addr[56:0], type=2
  u32x4 g0 = { 1u, lds_byte, (u32)ga, (u32)((ga >> 32) & 0x1FFFFFFull) | 0x80000000u };
  // D# group1: data_size=1 (2B), tensor_dim0=row_stride, tensor_dim1=1<<20,
  // tile_dim0=tile_k, tile_dim1=tile_rows, tensor_dim0_stride=row_stride
  i32x8 g1;
  g1[0] = (int)(1u << 16);
  g1[1] = (int)((row_stride_elems & 0xffffu) << 16);
  g1[2] = (int)((row_stride_elems >> 16) & 0xffffu);        // tensor_dim1 lo16 = 0
  g1[3] = (int)(16u | ((tile_k & 0xffffu) << 16));          // tensor_dim1 hi16 = 16 (=1M)
  g1[4] = (int)(tile_rows & 0xffffu);                       // tile_dim1; tile_dim2=0
  g1[5] = (int)row_stride_elems;                            // dim0 stride lo32
  g1[6] = 0;
  g1[7] = 0;
  i32x4 gz = { 0, 0, 0, 0 };
#if __has_include(<hip/amd_detail/amd_gfx1250_TDM.h>)
  i32x8 gz8 = { 0, 0, 0, 0, 0, 0, 0, 0 };
  __builtin_amdgcn_tensor_load_to_lds(g0, g1, gz, gz, gz8, 0);
#else
  __builtin_amdgcn_tensor_load_to_lds(g0, g1, gz, gz, 0);
#endif
#else
  (void)gsrc; (void)lds_byte; (void)tile_k; (void)tile_rows; (void)row_stride_elems;
#endif
}

__device__ __forceinline__ void wait_tensor0() {
#if HAVE_TDM
#if __has_builtin(__builtin_amdgcn_s_wait_tensorcnt)
  __builtin_amdgcn_s_wait_tensorcnt(0);
#else
  asm volatile("s_wait_tensorcnt 0x0" ::: "memory");
#endif
#endif
}

__device__ __forceinline__ u32 lds_off(const void* p) {
  return (u32)(unsigned long long)p;   // generic LDS addr low 32 bits = wave LDS offset
}

#if !HAVE_TDM
__device__ __forceinline__ void load_tile_fallback(const u16* gsrc, u16* dst, int tile_k,
                                                   int tile_rows, int row_stride,
                                                   int tid, int nthreads) {
  int kw = tile_k / 2;
  int total = kw * tile_rows;
  for (int i = tid; i < total; i += nthreads) {
    int rr = i / kw, cc = i % kw;
    ((u32*)dst)[i] = *(const u32*)(gsrc + (size_t)rr * row_stride + cc * 2);
  }
}
#endif

// ---------------- Generic K=512 bf16 WMMA GEMM: C[M,N] = A[M,512] * BT[N,512]^T ----
// Block: 256 threads (8 waves). Tile: 64(M) x 128(N). TDM double-buffered LDS staging.
__global__ __launch_bounds__(256) void gemm512_kernel(const u16* __restrict__ A,
                                                      const u16* __restrict__ BT,
                                                      const float* __restrict__ bias,
                                                      float* __restrict__ outF,
                                                      u16* __restrict__ outB, int N) {
  __shared__ u16 a_t[2][64 * 32];
  __shared__ u16 b_t[2][128 * 32];
  const int K = 512;
  const int m0 = blockIdx.x * 64;
  const int n0 = blockIdx.y * 128;
  const int tid = threadIdx.x, lane = tid & 31, wave = tid >> 5;
  const int wm = wave & 3;   // M tile 0..3
  const int wn = wave >> 2;  // N tile group 0..1 (4 tiles each)

  v8f acc[4] = { {}, {}, {}, {} };

#if HAVE_TDM
  if (tid < 32) {
    tdm_load_tile(A + (size_t)m0 * K, lds_off(a_t[0]), 32, 64, K);
    tdm_load_tile(BT + (size_t)n0 * K, lds_off(b_t[0]), 32, 128, K);
  }
#else
  load_tile_fallback(A + (size_t)m0 * K, a_t[0], 32, 64, K, tid, 256);
  load_tile_fallback(BT + (size_t)n0 * K, b_t[0], 32, 128, K, tid, 256);
#endif

  for (int kc = 0; kc < 16; ++kc) {
    const int cur = kc & 1;
#if HAVE_TDM
    if (tid < 32) wait_tensor0();
    __syncthreads();
    if (kc + 1 < 16 && tid < 32) {
      const int k1 = (kc + 1) * 32;
      tdm_load_tile(A + (size_t)m0 * K + k1, lds_off(a_t[cur ^ 1]), 32, 64, K);
      tdm_load_tile(BT + (size_t)n0 * K + k1, lds_off(b_t[cur ^ 1]), 32, 128, K);
    }
#else
    __syncthreads();
    if (kc + 1 < 16) {
      const int k1 = (kc + 1) * 32;
      load_tile_fallback(A + (size_t)m0 * K + k1, a_t[cur ^ 1], 32, 64, K, tid, 256);
      load_tile_fallback(BT + (size_t)n0 * K + k1, b_t[cur ^ 1], 32, 128, K, tid, 256);
    }
#endif
    v16bf af = load_a_frag(&a_t[cur][wm * 16 * 32], 32, 0, lane);
#pragma unroll
    for (int j = 0; j < 4; ++j) {
      v16bf bf_ = load_b_frag(&b_t[cur][0], 32, (wn * 4 + j) * 16, 0, lane);
      acc[j] = wmma_bf16(af, bf_, acc[j]);
    }
    asm volatile("s_wait_dscnt 0x0" ::: "memory");   // LDS reads done before DMA overwrite
    __syncthreads();
  }

  const int nl = lane & 15, rh = (lane >> 4) & 1;
#pragma unroll
  for (int j = 0; j < 4; ++j) {
    const int n = n0 + (wn * 4 + j) * 16 + nl;
    const float bv = bias ? bias[n] : 0.f;
#pragma unroll
    for (int r = 0; r < 8; ++r) {
      const int m = m0 + wm * 16 + r + 8 * rh;
      const float v = acc[j][r] + bv;
      if (outF) outF[(size_t)m * N + n] = v;
      else      outB[(size_t)m * N + n] = f2bf(v);
    }
  }
}

// ---------------- Global barrier (16 persistent WGs, monotonic slots) ----------------
__device__ __forceinline__ void gbar(int* bars, int slot) {
  __threadfence();
  __syncthreads();
  if (threadIdx.x == 0) {
    __hip_atomic_fetch_add(&bars[slot], 1, __ATOMIC_ACQ_REL, __HIP_MEMORY_SCOPE_AGENT);
    while (__hip_atomic_load(&bars[slot], __ATOMIC_ACQUIRE, __HIP_MEMORY_SCOPE_AGENT) <
           (int)gridDim.x) {
      __builtin_amdgcn_s_sleep(2);
    }
  }
  __syncthreads();
  __threadfence();
}

// ---------------- Persistent LSTM + Luong attention recurrence -----------------------
// 16 WGs x 256 threads. WG wg owns features [wg*32, wg*32+32): computes all 4 gates
// for them (z GEMM via WMMA), keeps c in LDS, and is also "batch wg" for attention.
__global__ __launch_bounds__(256) void decoder_recurrence_kernel(
    const u16* __restrict__ x_bf,    // [B*T][E]   embedded tokens, bf16
    const u16* __restrict__ WkrT,    // [2048][1280] = [Wk;Wr]^T bf16
    const float* __restrict__ bias,  // [2048]
    const u16* __restrict__ WaT,     // [512][1024] = Wa^T bf16
    const u16* __restrict__ keys,    // [B*S][F] bf16
    const float* __restrict__ memory,// [B][S][F] f32
    const float* __restrict__ c0,    // [B][F]
    u16* hc_g,                       // [B][1024]  (h | ctx) bf16
    u16* attn_g,                     // [B][F]     attention state bf16
    u16* attns,                      // [B*T][F]   per-step attention outputs bf16
    int* bars) {
  __shared__ u16   abuf[16][KZ];     // [x_t | attn | h] staged A operand (40 KB)
  __shared__ float zbuf[4][16][32];  // z tiles per gate
  __shared__ float cbuf[16][32];     // persistent cell state slice
  __shared__ float hvec[FF];
  __shared__ float red[256];
  __shared__ float align_s[SS];

  const int wg = blockIdx.x;
  const int tid = threadIdx.x;
  const int lane = tid & 31;
  const int wave = tid >> 5;

  for (int i = tid; i < 16 * 32; i += 256) {
    int b = i >> 5, j = i & 31;
    cbuf[b][j] = c0[b * FF + wg * 32 + j];
  }
  __syncthreads();

  for (int t = 0; t < TT; ++t) {
    // ---- stage A = [x_t | attn_{t-1} | h_{t-1}] into LDS (bf16, 16 x 1280) ----
    for (int i = tid; i < 16 * (KZ / 2); i += 256) {
      int b = i / (KZ / 2);
      int col = (i % (KZ / 2)) * 2;
      u32 v;
      if (col < EE)            v = *(const u32*)&x_bf[((size_t)(b * TT + t)) * EE + col];
      else if (col < EE + FF)  v = *(const u32*)&attn_g[b * FF + (col - EE)];
      else                     v = *(const u32*)&hc_g[b * 1024 + (col - (EE + FF))];
      *(u32*)&abuf[b][col] = v;
    }
    __syncthreads();

    // ---- z = A @ [Wk;Wr] + b : each wave one 16x16 tile (gate g, 16 cols) ----
    {
      const int g = wave >> 1;
      const int n0 = g * FF + wg * 32 + (wave & 1) * 16;
      v8f acc = {};
      for (int kc = 0; kc < KZ / 32; ++kc) {
        v16bf af  = load_a_frag(&abuf[0][0], KZ, kc * 32, lane);
        v16bf bf_ = load_b_frag(WkrT, KZ, n0, kc * 32, lane);
        acc = wmma_bf16(af, bf_, acc);
      }
      const float bv = bias[n0 + (lane & 15)];
      const int nl = (wave & 1) * 16 + (lane & 15);
#pragma unroll
      for (int r = 0; r < 8; ++r) {
        const int m = r + 8 * ((lane >> 4) & 1);
        zbuf[g][m][nl] = acc[r] + bv;
      }
    }
    __syncthreads();

    // ---- gates i,f,g,o -> c,h; publish h slice ----
    for (int i = tid; i < 16 * 32; i += 256) {
      int b = i >> 5, j = i & 31;
      float zi = zbuf[0][b][j], zf = zbuf[1][b][j], zg = zbuf[2][b][j], zo = zbuf[3][b][j];
      float c = sigmoidf(zf) * cbuf[b][j] + sigmoidf(zi) * tanhf(zg);
      cbuf[b][j] = c;
      hc_g[b * 1024 + wg * 32 + j] = f2bf(sigmoidf(zo) * tanhf(c));
    }
    gbar(bars, t * 3 + 0);

    // ---- Luong attention for batch b = wg ----
    {
      const int b = wg;
      for (int i = tid; i < FF; i += 256) hvec[i] = bf2f(hc_g[b * 1024 + i]);
      __syncthreads();
      {
        const int s = tid >> 1, half = tid & 1;
        const u16* kr = &keys[((size_t)(b * SS + s)) * FF + half * 256];
        float p = 0.f;
        for (int d = 0; d < 256; ++d) p += hvec[half * 256 + d] * bf2f(kr[d]);
        red[tid] = p;
      }
      __syncthreads();
      if (tid < SS) align_s[tid] = red[2 * tid] + red[2 * tid + 1];
      __syncthreads();
      if (tid < 64) red[tid] = fmaxf(align_s[tid], align_s[tid + 64]);
      __syncthreads();
      for (int off = 32; off >= 1; off >>= 1) {
        if (tid < off) red[tid] = fmaxf(red[tid], red[tid + off]);
        __syncthreads();
      }
      const float mx = red[0];
      __syncthreads();
      if (tid < SS) align_s[tid] = __expf(align_s[tid] - mx);
      __syncthreads();
      if (tid < 64) red[tid] = align_s[tid] + align_s[tid + 64];
      __syncthreads();
      for (int off = 32; off >= 1; off >>= 1) {
        if (tid < off) red[tid] += red[tid + off];
        __syncthreads();
      }
      const float inv = 1.f / red[0];
      __syncthreads();
      if (tid < SS) align_s[tid] *= inv;
      __syncthreads();
      // ctx[d] = sum_s align[s] * memory[b,s,d]  (coalesced across threads per s)
      for (int base = 0; base < FF; base += 256) {
        const int d = base + tid;
        float a = 0.f;
        for (int s = 0; s < SS; ++s) a += align_s[s] * memory[((size_t)(b * SS + s)) * FF + d];
        hc_g[b * 1024 + FF + d] = f2bf(a);
      }
    }
    gbar(bars, t * 3 + 1);

    // ---- attn = [h|ctx] @ Wa : WG's 32 columns, waves 0..1, K=1024 ----
    if (wave < 2) {
      const int n0a = wg * 32 + wave * 16;
      v8f acc = {};
      for (int kc = 0; kc < 1024 / 32; ++kc) {
        v16bf af  = load_a_frag(hc_g, 1024, kc * 32, lane);
        v16bf bf_ = load_b_frag(WaT, 1024, n0a, kc * 32, lane);
        acc = wmma_bf16(af, bf_, acc);
      }
      const int nl = lane & 15, rh = (lane >> 4) & 1;
#pragma unroll
      for (int r = 0; r < 8; ++r) {
        const int m = r + 8 * rh;  // batch
        const u16 v = f2bf(acc[r]);
        attn_g[m * FF + n0a + nl] = v;
        attns[((size_t)(m * TT + t)) * FF + n0a + nl] = v;
      }
    }
    gbar(bars, t * 3 + 2);
  }
}

// ---------------- Data preparation kernels ----------------
__global__ void cast_bf16_kernel(const float* __restrict__ src, u16* __restrict__ dst, int n) {
  int i = blockIdx.x * 256 + threadIdx.x;
  if (i < n) dst[i] = f2bf(src[i]);
}

__global__ void transpose_bf16_kernel(const float* __restrict__ src /*[K][N]*/,
                                      u16* __restrict__ dst /*[N][K]*/, int K, int N) {
  long long i = (long long)blockIdx.x * 256 + threadIdx.x;
  long long total = (long long)K * N;
  if (i < total) {
    int k = (int)(i / N), n = (int)(i % N);
    dst[(long long)n * K + k] = f2bf(src[i]);
  }
}

__global__ void wkr_transpose_kernel(const float* __restrict__ Wk, const float* __restrict__ Wr,
                                     u16* __restrict__ dst /*[2048][1280]*/) {
  long long i = (long long)blockIdx.x * 256 + threadIdx.x;
  if (i < (long long)NZ * KZ) {
    int n = (int)(i / KZ), k = (int)(i % KZ);
    float v = (k < EE + FF) ? Wk[(long long)k * NZ + n] : Wr[(long long)(k - (EE + FF)) * NZ + n];
    dst[i] = f2bf(v);
  }
}

__global__ void gather_emb_kernel(const int* __restrict__ tokens, const float* __restrict__ emb,
                                  u16* __restrict__ x_bf) {
  int i = blockIdx.x * 256 + threadIdx.x;
  if (i < BB * TT * EE) {
    int row = i >> 8, e = i & 255;
    int tok = tokens[row];
    x_bf[i] = f2bf(emb[(long long)tok * EE + e]);
  }
}

__global__ void init_state_kernel(const float* __restrict__ h0, u16* hc_g, u16* attn_g, int* bars) {
  int i = blockIdx.x * 256 + threadIdx.x;
  if (i < BB * 1024) {
    int b = i >> 10, c = i & 1023;
    hc_g[i] = (c < FF) ? f2bf(h0[b * FF + c]) : (u16)0;
  }
  if (i < BB * FF) attn_g[i] = 0;
  if (i < 3 * TT) bars[i] = 0;
}

// ---------------- Host orchestration ----------------
extern "C" void kernel_launch(void* const* d_in, const int* in_sizes, int n_in,
                              void* d_out, int out_size, void* d_ws, size_t ws_size,
                              hipStream_t stream) {
  (void)in_sizes; (void)n_in; (void)out_size; (void)ws_size;
  const int*   tokens = (const int*)  d_in[0];
  const float* memory = (const float*)d_in[1];
  const float* h0     = (const float*)d_in[2];
  const float* c0     = (const float*)d_in[3];
  const float* emb    = (const float*)d_in[4];
  const float* Wk     = (const float*)d_in[5];
  const float* Wr     = (const float*)d_in[6];
  const float* bvec   = (const float*)d_in[7];
  const float* Wm     = (const float*)d_in[8];
  const float* Wa     = (const float*)d_in[9];
  const float* Wf     = (const float*)d_in[10];
  const float* bf     = (const float*)d_in[11];
  float* out = (float*)d_out;

  char* ws = (char*)d_ws;
  size_t off = 0;
  auto carve = [&](size_t bytes) -> void* {
    off = (off + 255) & ~(size_t)255;
    void* p = ws + off;
    off += bytes;
    return p;
  };
  u16* mem_bf  = (u16*)carve((size_t)BB * SS * FF * 2);   // 2 MB
  u16* WmT     = (u16*)carve((size_t)FF * FF * 2);        // 512 KB
  u16* keys_bf = (u16*)carve((size_t)BB * SS * FF * 2);   // 2 MB
  u16* x_bf    = (u16*)carve((size_t)BB * TT * EE * 2);   // 1 MB
  u16* WkrT    = (u16*)carve((size_t)NZ * KZ * 2);        // 5.25 MB
  u16* WaT     = (u16*)carve((size_t)FF * 1024 * 2);      // 1 MB
  u16* WfT     = (u16*)carve((size_t)VV * FF * 2);        // 32.8 MB
  u16* hc_g    = (u16*)carve((size_t)BB * 1024 * 2);
  u16* attn_g  = (u16*)carve((size_t)BB * FF * 2);
  u16* attns   = (u16*)carve((size_t)BB * TT * FF * 2);   // 2 MB
  int* bars    = (int*)carve((size_t)3 * TT * 4);

  dim3 blk(256);
  cast_bf16_kernel<<<(BB * SS * FF + 255) / 256, blk, 0, stream>>>(memory, mem_bf, BB * SS * FF);
  transpose_bf16_kernel<<<(FF * FF + 255) / 256, blk, 0, stream>>>(Wm, WmT, FF, FF);
  transpose_bf16_kernel<<<(1024 * FF + 255) / 256, blk, 0, stream>>>(Wa, WaT, 1024, FF);
  transpose_bf16_kernel<<<(FF * VV + 255) / 256, blk, 0, stream>>>(Wf, WfT, FF, VV);
  wkr_transpose_kernel<<<(NZ * KZ + 255) / 256, blk, 0, stream>>>(Wk, Wr, WkrT);
  gather_emb_kernel<<<(BB * TT * EE + 255) / 256, blk, 0, stream>>>(tokens, emb, x_bf);
  init_state_kernel<<<(BB * 1024 + 255) / 256, blk, 0, stream>>>(h0, hc_g, attn_g, bars);

  // keys = memory @ Wm  -> bf16 [B*S, F]
  gemm512_kernel<<<dim3((BB * SS) / 64, FF / 128), blk, 0, stream>>>(
      mem_bf, WmT, nullptr, nullptr, keys_bf, FF);

  // sequential decoder recurrence (16 persistent WGs)
  decoder_recurrence_kernel<<<16, blk, 0, stream>>>(
      x_bf, WkrT, bvec, WaT, keys_bf, memory, c0, hc_g, attn_g, attns, bars);

  // logits[b*T+t, :] = attns @ Wf + bf  (fp32 out, row id matches [B,T,V] layout)
  gemm512_kernel<<<dim3((BB * TT) / 64, VV / 128), blk, 0, stream>>>(
      attns, WfT, bf, out, nullptr, VV);
}
